// TransformerEncoder_24979529793753
// MI455X (gfx1250) — matrix operand order repaired and checked by reading, not compiled, not run on hardware
//
#include <hip/hip_runtime.h>

// ---------------- problem constants ----------------
constexpr int kL = 4, kH = 8, kE = 512, kFF = 2048, kMaxP = 10, kNK = 1000;
constexpr int kB = 4, kS = 128, kHD = 64, kBS = kB * kS;

// ---------------- WMMA plumbing (gfx1250, wave32) ----------------
typedef __attribute__((ext_vector_type(16))) __bf16 v16bf;
typedef __attribute__((ext_vector_type(8)))  float  v8f;

union FragB { v16bf v; uint4 q[2]; };

__device__ inline v8f wmma_bf16f32(v16bf a, v16bf b, v8f c) {
  // D = A(16x32 bf16) x B(32x16 bf16) + C(16x16 f32)
  return __builtin_amdgcn_wmma_f32_16x16x32_bf16(false, a, false, b, (short)0, c,
                                                 false, false);
}

__device__ inline v8f v8f_zero() {
  v8f r;
#pragma unroll
  for (int i = 0; i < 8; ++i) r[i] = 0.0f;
  return r;
}

__device__ inline __bf16 f2bf(float f) {          // round-to-nearest-even
  unsigned u = __float_as_uint(f);
  u += 0x7FFFu + ((u >> 16) & 1u);
  unsigned short h = (unsigned short)(u >> 16);
  __bf16 r; __builtin_memcpy(&r, &h, 2); return r;
}

// ---------------- CDNA5 async global->LDS copy (ASYNCcnt-tracked) ----------
// LDS byte address = low 32 bits of the generic pointer (ISA 10.2: flat/LDS
// addresses truncate to addr[31:0]).
__device__ inline unsigned lds_addr(const void* p) {
  return (unsigned)(unsigned long long)(uintptr_t)p;
}
__device__ inline void async_copy_b128(unsigned lds_off, const void* gaddr) {
  asm volatile("global_load_async_to_lds_b128 %0, %1, off"
               :: "v"(lds_off), "v"(gaddr) : "memory");
}
template <int N>
__device__ inline void wait_async_le() {
  asm volatile("s_wait_asynccnt %0" :: "i"(N) : "memory");
}

// ---------------- elementwise converts ----------------
__global__ __launch_bounds__(256) void f32_to_bf16_kernel(const float* __restrict__ in,
                                                          __bf16* __restrict__ out,
                                                          long long n) {
  long long i = (long long)blockIdx.x * 256 + threadIdx.x;
  long long stride = (long long)gridDim.x * 256;
  for (; i < n; i += stride) out[i] = f2bf(in[i]);
}

__global__ __launch_bounds__(256) void init_h_kernel(const float* __restrict__ x,
                                                     float* __restrict__ h,
                                                     __bf16* __restrict__ hB,
                                                     long long n) {
  long long i = (long long)blockIdx.x * 256 + threadIdx.x;
  long long stride = (long long)gridDim.x * 256;
  for (; i < n; i += stride) { float v = x[i]; h[i] = v; hB[i] = f2bf(v); }
}

// ---------------- generic bf16 WMMA GEMM: C[m,n] = sum_k A[m,k] * B(k,n) ----
// transB=1: B stored [N,K] row-major (x @ W.T case).  transB=0: B stored [K,N].
// flags: 1=bias, 2=relu, 4=store f32 C, 8=store bf16 Cb
// Double-buffered LDS tiles filled with global_load_async_to_lds_b128.
__global__ __launch_bounds__(256) void gemm_bf16_kernel(
    const __bf16* __restrict__ A, int lda, long long sAz,
    const __bf16* __restrict__ Bm, int ldb, long long sBz,
    float* __restrict__ C, __bf16* __restrict__ Cb, int ldc, long long sCz,
    const float* __restrict__ bias, int K, int flags, int transB) {
  constexpr int BM = 128, BN = 64, BK = 32, BKP = BK + 8;
  __shared__ __bf16 As[2][BM][BKP];   // pitch 80B (16B aligned, conflict-free)
  __shared__ __bf16 Bs[2][BN][BKP];

  int z = blockIdx.z;
  A  += (long long)z * sAz;
  Bm += (long long)z * sBz;
  long long cOff = (long long)z * sCz;
  int mBase = blockIdx.x * BM, nBase = blockIdx.y * BN;
  int tid = threadIdx.x;
  int wave = tid >> 5, lane = tid & 31, half = lane >> 4, ln = lane & 15;
  int wm = wave >> 1, wn = wave & 1;          // 4x2 wave grid -> 32x32 per wave

  v8f acc[2][2];
#pragma unroll
  for (int i = 0; i < 2; ++i)
#pragma unroll
    for (int j = 0; j < 2; ++j) acc[i][j] = v8f_zero();

  auto issueTile = [&](int kb, int buf) {
    int k0 = kb * BK;
    // A tile: 128x32 bf16, 512 16B chunks -> 2 async copies per thread
    for (int c = tid; c < BM * BK / 8; c += 256) {
      int r = c >> 2, cc = (c & 3) * 8;
      async_copy_b128(lds_addr(&As[buf][r][cc]),
                      (const void*)&A[(long long)(mBase + r) * lda + k0 + cc]);
    }
    if (transB) {                              // 1 async copy per thread
      for (int c = tid; c < BN * BK / 8; c += 256) {
        int r = c >> 2, cc = (c & 3) * 8;
        async_copy_b128(lds_addr(&Bs[buf][r][cc]),
                        (const void*)&Bm[(long long)(nBase + r) * ldb + k0 + cc]);
      }
    } else {                                   // transpose-on-store (qkn only)
      for (int c = tid; c < BN * BK; c += 256) {
        int n = c & 63, kk = c >> 6;
        Bs[buf][n][kk] = Bm[(long long)(k0 + kk) * ldb + nBase + n];
      }
    }
  };

  int nk = K / BK;
  issueTile(0, 0);
  for (int kb = 0; kb < nk; ++kb) {
    int buf = kb & 1;
    if (kb + 1 < nk) {
      issueTile(kb + 1, buf ^ 1);              // prefetch next tile
      // async ops complete in order: <=N outstanding => current tile done
      if (transB) wait_async_le<3>(); else wait_async_le<2>();
    } else {
      wait_async_le<0>();
    }
    __syncthreads();

    FragB af[2], bfr[2];
#pragma unroll
    for (int i = 0; i < 2; ++i) {              // A frag: K = 16*half + e
      int row = (wm * 2 + i) * 16 + ln;
      af[i].q[0] = *(const uint4*)&As[buf][row][16 * half];
      af[i].q[1] = *(const uint4*)&As[buf][row][16 * half + 8];
    }
#pragma unroll
    for (int j = 0; j < 2; ++j) {              // B frag: K = 16*half + e
      int row = (wn * 2 + j) * 16 + ln;
      bfr[j].q[0] = *(const uint4*)&Bs[buf][row][16 * half];
      bfr[j].q[1] = *(const uint4*)&Bs[buf][row][16 * half + 8];
    }
#pragma unroll
    for (int i = 0; i < 2; ++i)
#pragma unroll
      for (int j = 0; j < 2; ++j)
        acc[i][j] = wmma_bf16f32(af[i].v, bfr[j].v, acc[i][j]);
    __syncthreads();
  }

#pragma unroll
  for (int i = 0; i < 2; ++i)
#pragma unroll
    for (int j = 0; j < 2; ++j) {
      int nG = nBase + (wn * 2 + j) * 16 + ln;
      float bv = (flags & 1) ? bias[nG] : 0.0f;
#pragma unroll
      for (int v = 0; v < 8; ++v) {            // C map: m = v + 8*half, n = ln
        int mG = mBase + (wm * 2 + i) * 16 + v + half * 8;
        float val = acc[i][j][v] + bv;
        if (flags & 2) val = fmaxf(val, 0.0f);
        long long idx = cOff + (long long)mG * ldc + nG;
        if (flags & 4) C[idx] = val;
        if (flags & 8) Cb[idx] = f2bf(val);
      }
    }
}

// ------- knowledge scores: sk[b,h,i,j] = qproj[b,i,h,:] . kn[know_adj[b,i,j],:]
// one WG per (b,i); wave w owns j-tile w; K=512 -> 16 wmma steps.
__global__ __launch_bounds__(256) void know_scores_kernel(
    const __bf16* __restrict__ qp,   // [BS, H*E]
    const __bf16* __restrict__ knB,  // [NK, E]
    const int* __restrict__ kadj,    // [B,S,S]
    float* __restrict__ sk) {        // [B,H,S,S]
  __shared__ __bf16 qpl[kH * kE];    // 8KB: the 8 qproj rows for this (b,i)
  __shared__ int rows[kS];
  int bi = blockIdx.x;
  int t = threadIdx.x;
  const __bf16* qrow = qp + (long long)bi * (kH * kE);
  for (int c = t; c < (kH * kE) / 8; c += 256)            // 2 async per thread
    async_copy_b128(lds_addr(&qpl[c * 8]), (const void*)&qrow[c * 8]);
  if (t < kS) {
    int r = kadj[(long long)bi * kS + t];
    rows[t] = r < 0 ? 0 : (r >= kNK ? kNK - 1 : r);
  }
  wait_async_le<0>();
  __syncthreads();

  int wave = t >> 5, lane = t & 31, half = lane >> 4, ln = lane & 15;
  int j = wave * 16 + ln;
  const __bf16* krow = knB + (long long)rows[j] * kE;  // L2-resident gather
  __builtin_prefetch(krow, 0, 1);
  const __bf16* arow = &qpl[(ln & 7) * kE];  // rows 8..15 duplicate (not stored)
  v8f acc = v8f_zero();
  for (int k0 = 0; k0 < kE; k0 += 32) {
    FragB a, bf;
    a.q[0]  = *(const uint4*)&arow[k0 + 16 * half];
    a.q[1]  = *(const uint4*)&arow[k0 + 16 * half + 8];
    bf.q[0] = *(const uint4*)&krow[k0 + 16 * half];
    bf.q[1] = *(const uint4*)&krow[k0 + 16 * half + 8];
    acc = wmma_bf16f32(a.v, bf.v, acc);
  }
  if (half == 0) {                            // m = v (h index), valid only m<8
    int b = bi / kS, i = bi % kS;
#pragma unroll
    for (int v = 0; v < 8; ++v)
      sk[(((long long)(b * kH + v)) * kS + i) * kS + j] = acc[v];
  }
}

// ------- fused scores + softmax: one WG of 4 waves per (b,h); wave owns row i
__global__ __launch_bounds__(128) void scores_softmax_kernel(
    const float* __restrict__ qF, const float* __restrict__ kF,
    const float* __restrict__ sk, const float* __restrict__ peK,
    const int* __restrict__ posm, const int* __restrict__ adj,
    const int* __restrict__ smask, const int* __restrict__ omask,
    __bf16* __restrict__ aws, __bf16* __restrict__ awo,
    float* __restrict__ apos) {
  __shared__ float qlds[kS][kHD + 1];
  __shared__ float klds[kS][kHD + 1];
  __shared__ float qpeL[kS][12];
  __shared__ float pek[11][kHD];
  __shared__ float bins[4][16];
  int bh = blockIdx.x, b = bh / kH, h = bh % kH;
  int t = threadIdx.x;
  for (int idx = t; idx < kS * kHD; idx += 128) {
    int r = idx >> 6, d = idx & 63;
    long long gi = ((long long)(b * kS + r)) * kE + h * kHD + d;
    qlds[r][d] = qF[gi];
    klds[r][d] = kF[gi];
  }
  for (int idx = t; idx < 11 * kHD; idx += 128) pek[idx >> 6][idx & 63] = peK[idx];
  __syncthreads();
  for (int idx = t; idx < kS * 11; idx += 128) {  // qpe[i][p] = q_i . pe_k[p]
    int i = idx / 11, p = idx % 11;
    float s = 0.f;
#pragma unroll 8
    for (int d = 0; d < kHD; ++d) s += qlds[i][d] * pek[p][d];
    qpeL[i][p] = s;
  }
  __syncthreads();

  int wave = t >> 5, lane = t & 31;
  const float scaling = 0.125f;  // hd^-0.5
  for (int i0 = 0; i0 < kS; i0 += 4) {
    int i = i0 + wave;
    const float* skrow = sk + ((long long)bh * kS + i) * kS;
    const int* adjr = adj + ((long long)(b * kS + i)) * kS;
    const int* posr = posm + i * kS;
    const int* smr = smask + ((long long)(b * kS + i)) * kS;
    const int* omr = omask + ((long long)(b * kS + i)) * kS;
    float sc4[4], e4[4]; int p4[4];
    float mx = -3.0e38f;
#pragma unroll
    for (int tt = 0; tt < 4; ++tt) {
      int j = lane + 32 * tt;
      float d = 0.f;
#pragma unroll 8
      for (int dd = 0; dd < kHD; ++dd) d += qlds[i][dd] * klds[j][dd];
      int p = posr[j]; p = p < 0 ? 0 : (p > kMaxP ? kMaxP : p);
      p4[tt] = p;
      float s = (d + skrow[j] + qpeL[i][p]) * scaling;
      s -= (1.0f - (float)adjr[j]) * 1e30f;
      sc4[tt] = s; mx = fmaxf(mx, s);
    }
#pragma unroll
    for (int off = 16; off > 0; off >>= 1) mx = fmaxf(mx, __shfl_xor(mx, off));
    float sum = 0.f;
#pragma unroll
    for (int tt = 0; tt < 4; ++tt) { e4[tt] = __expf(sc4[tt] - mx); sum += e4[tt]; }
#pragma unroll
    for (int off = 16; off > 0; off >>= 1) sum += __shfl_xor(sum, off);
    float inv = 1.0f / sum;
    if (lane < 11) bins[wave][lane] = 0.f;     // per-wave LDS ops are in-order
#pragma unroll
    for (int tt = 0; tt < 4; ++tt) {
      int j = lane + 32 * tt;
      float a = e4[tt] * inv;
      long long oidx = ((long long)bh * kS + i) * kS + j;
      aws[oidx] = f2bf(a * (float)smr[j]);
      awo[oidx] = f2bf(a * (float)omr[j]);
      atomicAdd(&bins[wave][p4[tt]], a);
    }
    if (lane < 11) apos[((long long)bh * kS + i) * 11 + lane] = bins[wave][lane];
  }
}

// ------- attention apply: out = aws@vs + awo@vo + attn_pos@pe_v (per (b,h))
__global__ __launch_bounds__(256) void attn_apply_kernel(
    const __bf16* __restrict__ aws, const __bf16* __restrict__ awo,
    const __bf16* __restrict__ vsB, const __bf16* __restrict__ voB,
    const float* __restrict__ apos, const float* __restrict__ peV,
    __bf16* __restrict__ outB) {
  __shared__ __bf16 vsT[kHD][kS + 8];   // transposed (d-major) for B-fragments
  __shared__ __bf16 voT[kHD][kS + 8];
  __shared__ float apl[kS][12];
  __shared__ float pev[11][kHD];
  int bh = blockIdx.x, b = bh >> 3, h = bh & 7;
  int t = threadIdx.x;
  union { uint4 q; __bf16 e[8]; } tmp;
  for (int c = t; c < kS * kHD / 8; c += 256) {
    int j = c >> 3, d0 = (c & 7) * 8;
    long long gi = ((long long)(b * kS + j)) * kE + h * kHD + d0;
    tmp.q = *(const uint4*)&vsB[gi];
#pragma unroll
    for (int e = 0; e < 8; ++e) vsT[d0 + e][j] = tmp.e[e];
    tmp.q = *(const uint4*)&voB[gi];
#pragma unroll
    for (int e = 0; e < 8; ++e) voT[d0 + e][j] = tmp.e[e];
  }
  for (int idx = t; idx < kS * 11; idx += 256)
    apl[idx / 11][idx % 11] = apos[(long long)bh * kS * 11 + idx];
  for (int idx = t; idx < 11 * kHD; idx += 256) pev[idx >> 6][idx & 63] = peV[idx];
  __syncthreads();

  int wave = t >> 5, lane = t & 31, half = lane >> 4, ln = lane & 15;
  v8f acc[4];
#pragma unroll
  for (int i = 0; i < 4; ++i) acc[i] = v8f_zero();
  const __bf16* awsRow = aws + ((long long)bh * kS + wave * 16 + ln) * kS;
  const __bf16* awoRow = awo + ((long long)bh * kS + wave * 16 + ln) * kS;
#pragma unroll
  for (int pass = 0; pass < 2; ++pass) {      // chained GEMMs, same accumulators
    const __bf16* ar = pass ? awoRow : awsRow;
    const __bf16* vt = pass ? &voT[0][0] : &vsT[0][0];
    for (int k0 = 0; k0 < kS; k0 += 32) {
      FragB a;
      a.q[0] = *(const uint4*)&ar[k0 + 16 * half];
      a.q[1] = *(const uint4*)&ar[k0 + 16 * half + 8];
#pragma unroll
      for (int dt = 0; dt < 4; ++dt) {
        FragB bf;
        const __bf16* br = vt + (dt * 16 + ln) * (kS + 8);
        bf.q[0] = *(const uint4*)&br[k0 + 16 * half];
        bf.q[1] = *(const uint4*)&br[k0 + 16 * half + 8];
        acc[dt] = wmma_bf16f32(a.v, bf.v, acc[dt]);
      }
    }
  }
#pragma unroll
  for (int dt = 0; dt < 4; ++dt) {
    int d = dt * 16 + ln;
#pragma unroll
    for (int v = 0; v < 8; ++v) {
      int i = wave * 16 + v + half * 8;
      float rv = 0.f;
#pragma unroll
      for (int p = 0; p < 11; ++p) rv += apl[i][p] * pev[p][d];
      outB[((long long)(b * kS + i)) * kE + h * kHD + d] = f2bf(acc[dt][v] + rv);
    }
  }
}

// ------- residual + LayerNorm, writes f32 (+ optional bf16 for next GEMMs)
__global__ __launch_bounds__(256) void add_ln_kernel(
    const float* __restrict__ hin, const float* __restrict__ res,
    const float* __restrict__ g, const float* __restrict__ bt,
    float* __restrict__ outF, __bf16* __restrict__ outB, int writeB) {
  __shared__ float w1s[8], w2s[8];
  int row = blockIdx.x, t = threadIdx.x;
  long long base = (long long)row * kE;
  float v0 = hin[base + t] + res[base + t];
  float v1 = hin[base + t + 256] + res[base + t + 256];
  float s = v0 + v1, ss = v0 * v0 + v1 * v1;
#pragma unroll
  for (int off = 16; off > 0; off >>= 1) { s += __shfl_xor(s, off); ss += __shfl_xor(ss, off); }
  int wave = t >> 5, lane = t & 31;
  if (lane == 0) { w1s[wave] = s; w2s[wave] = ss; }
  __syncthreads();
  float tot = 0.f, tot2 = 0.f;
#pragma unroll
  for (int w = 0; w < 8; ++w) { tot += w1s[w]; tot2 += w2s[w]; }
  float mean = tot * (1.0f / kE);
  float var = tot2 * (1.0f / kE) - mean * mean;
  float rstd = rsqrtf(var + 1e-5f);
  float y0 = (v0 - mean) * rstd * g[t] + bt[t];
  float y1 = (v1 - mean) * rstd * g[t + 256] + bt[t + 256];
  outF[base + t] = y0; outF[base + t + 256] = y1;
  if (writeB) { outB[base + t] = f2bf(y0); outB[base + t + 256] = f2bf(y1); }
}

// ---------------- host orchestration ----------------
extern "C" void kernel_launch(void* const* d_in, const int* in_sizes, int n_in,
                              void* d_out, int out_size, void* d_ws, size_t ws_size,
                              hipStream_t stream) {
  (void)in_sizes; (void)n_in; (void)out_size; (void)ws_size;
  const float* x         = (const float*)d_in[0];
  const int*   adj       = (const int*)d_in[1];
  const int*   smask     = (const int*)d_in[2];
  const int*   omask     = (const int*)d_in[3];
  const float* knowledge = (const float*)d_in[4];
  const int*   kadj      = (const int*)d_in[5];
  const int*   posm      = (const int*)d_in[6];
  const float* peK       = (const float*)d_in[7];
  const float* peV       = (const float*)d_in[8];
  const float* Wq  = (const float*)d_in[9];
  const float* Wk  = (const float*)d_in[10];
  const float* Wvs = (const float*)d_in[11];
  const float* Wvo = (const float*)d_in[12];
  const float* Wo  = (const float*)d_in[13];
  const float* Wkn = (const float*)d_in[14];
  const float* ln1g = (const float*)d_in[15];
  const float* ln1b = (const float*)d_in[16];
  const float* w1   = (const float*)d_in[17];
  const float* b1   = (const float*)d_in[18];
  const float* w2   = (const float*)d_in[19];
  const float* b2   = (const float*)d_in[20];
  const float* ln2g = (const float*)d_in[21];
  const float* ln2b = (const float*)d_in[22];

  char* p = (char*)d_ws;
  auto alloc = [&](size_t bytes) -> void* {
    void* r = (void*)p; p += (bytes + 255) & ~(size_t)255; return r;
  };
  const size_t WE = (size_t)kL * kE * kE;
  __bf16* WqB  = (__bf16*)alloc(WE * 2);
  __bf16* WkB  = (__bf16*)alloc(WE * 2);
  __bf16* WvsB = (__bf16*)alloc(WE * 2);
  __bf16* WvoB = (__bf16*)alloc(WE * 2);
  __bf16* WoB  = (__bf16*)alloc(WE * 2);
  __bf16* WknB = (__bf16*)alloc(WE * 2);
  __bf16* W1B  = (__bf16*)alloc((size_t)kL * kFF * kE * 2);
  __bf16* W2B  = (__bf16*)alloc((size_t)kL * kE * kFF * 2);
  __bf16* knB  = (__bf16*)alloc((size_t)kNK * kE * 2);
  float*  h    = (float*)alloc((size_t)kBS * kE * 4);
  __bf16* hB   = (__bf16*)alloc((size_t)kBS * kE * 2);
  float*  qF   = (float*)alloc((size_t)kBS * kE * 4);
  __bf16* qB   = (__bf16*)alloc((size_t)kBS * kE * 2);
  float*  kFb  = (float*)alloc((size_t)kBS * kE * 4);
  __bf16* vsB  = (__bf16*)alloc((size_t)kBS * kE * 2);
  __bf16* voB  = (__bf16*)alloc((size_t)kBS * kE * 2);
  __bf16* qpB  = (__bf16*)alloc((size_t)kBS * kH * kE * 2);
  float*  sk   = (float*)alloc((size_t)kB * kH * kS * kS * 4);
  __bf16* awsB = (__bf16*)alloc((size_t)kB * kH * kS * kS * 2);
  __bf16* awoB = (__bf16*)alloc((size_t)kB * kH * kS * kS * 2);
  float*  apos = (float*)alloc((size_t)kB * kH * kS * 11 * 4);
  __bf16* aoutB= (__bf16*)alloc((size_t)kBS * kE * 2);
  float*  pout = (float*)alloc((size_t)kBS * kE * 4);
  __bf16* ff1B = (__bf16*)alloc((size_t)kBS * kFF * 2);

  auto conv = [&](const float* src, __bf16* dst, long long n) {
    f32_to_bf16_kernel<<<1024, 256, 0, stream>>>(src, dst, n);
  };
  conv(Wq, WqB, WE);  conv(Wk, WkB, WE);  conv(Wvs, WvsB, WE);
  conv(Wvo, WvoB, WE); conv(Wo, WoB, WE); conv(Wkn, WknB, WE);
  conv(w1, W1B, (long long)kL * kFF * kE);
  conv(w2, W2B, (long long)kL * kE * kFF);
  conv(knowledge, knB, (long long)kNK * kE);
  init_h_kernel<<<1024, 256, 0, stream>>>(x, h, hB, (long long)kBS * kE);

  auto gemm = [&](const __bf16* A, int lda, long long sAz,
                  const __bf16* Bm, int ldb, long long sBz,
                  float* C, __bf16* Cb, int ldc, long long sCz,
                  const float* bias, int N, int K, int flags, int transB, int z) {
    dim3 g(kBS / 128, N / 64, z);
    gemm_bf16_kernel<<<g, 256, 0, stream>>>(A, lda, sAz, Bm, ldb, sBz,
                                            C, Cb, ldc, sCz, bias, K, flags, transB);
  };

  for (int l = 0; l < kL; ++l) {
    const __bf16* wq  = WqB  + (size_t)l * kE * kE;
    const __bf16* wk  = WkB  + (size_t)l * kE * kE;
    const __bf16* wvs = WvsB + (size_t)l * kE * kE;
    const __bf16* wvo = WvoB + (size_t)l * kE * kE;
    const __bf16* wo  = WoB  + (size_t)l * kE * kE;
    const __bf16* wkn = WknB + (size_t)l * kE * kE;

    // projections (x @ W.T): q needs f32 (scores) + bf16 (qkn GEMM)
    gemm(hB, kE, 0, wq,  kE, 0, qF, qB,      kE, 0, nullptr, kE, kE, 4 | 8, 1, 1);
    gemm(hB, kE, 0, wk,  kE, 0, kFb, nullptr, kE, 0, nullptr, kE, kE, 4,     1, 1);
    gemm(hB, kE, 0, wvs, kE, 0, nullptr, vsB, kE, 0, nullptr, kE, kE, 8,     1, 1);
    gemm(hB, kE, 0, wvo, kE, 0, nullptr, voB, kE, 0, nullptr, kE, kE, 8,     1, 1);
    // qproj[b,i,h,:] = q[b,i,h*64:+64] @ Wkn[h*64:+64, :]  (z = head, B is [K,N])
    gemm(qB, kE, kHD, wkn, kE, (long long)kHD * kE,
         nullptr, qpB, kH * kE, kE, nullptr, kE, kHD, 8, 0, kH);
    // knowledge gather-GEMM scores
    know_scores_kernel<<<kBS, 256, 0, stream>>>(qpB, knB, kadj, sk);
    // fused qk + sk + qpe + mask + softmax (+pos-bucketed attn sums)
    scores_softmax_kernel<<<kB * kH, 128, 0, stream>>>(
        qF, kFb, sk, peK, posm, adj, smask, omask, awsB, awoB, apos);
    // attn_sum = aws@vs + awo@vo + attn_pos@pe_v
    attn_apply_kernel<<<kB * kH, 256, 0, stream>>>(awsB, awoB, vsB, voB, apos, peV, aoutB);
    // output projection, residual + LN1
    gemm(aoutB, kE, 0, wo, kE, 0, pout, nullptr, kE, 0, nullptr, kE, kE, 4, 1, 1);
    add_ln_kernel<<<kBS, 256, 0, stream>>>(h, pout, ln1g + l * kE, ln1b + l * kE, h, hB, 1);
    // FFN
    gemm(hB, kE, 0, W1B + (size_t)l * kFF * kE, kE, 0,
         nullptr, ff1B, kFF, 0, b1 + (size_t)l * kFF, kFF, kE, 1 | 2 | 8, 1, 1);
    gemm(ff1B, kFF, 0, W2B + (size_t)l * kE * kFF, kFF, 0,
         pout, nullptr, kE, 0, b2 + (size_t)l * kE, kE, kFF, 1 | 4, 1, 1);
    int last = (l == kL - 1);
    add_ln_kernel<<<kBS, 256, 0, stream>>>(h, pout, ln2g + l * kE, ln2b + l * kE,
                                           last ? (float*)d_out : h, hB, last ? 0 : 1);
  }
}
